// ResidualBlock_68564857913684
// MI455X (gfx1250) — compile-verified, matrix operand order checked by reading, not run on hardware
//
#include <hip/hip_runtime.h>
#include <hip/hip_bf16.h>
#include <math.h>

typedef __bf16 bf16_t;
typedef __attribute__((ext_vector_type(16))) __bf16 v16bf;
typedef __attribute__((ext_vector_type(8)))  __bf16 v8bf;
typedef __attribute__((ext_vector_type(8)))  float  v8f;

namespace {

constexpr int B_ = 2, L_ = 512, M_ = B_ * L_;
constexpr int IN_CH = 64, RES_CH = 256, DIFF_DIM = 128;
constexpr int NSTATE = 16;
constexpr float SQRT_HALF = 0.70710678118654752440f;

__device__ __forceinline__ float siluf(float v) {
  return v * (1.f / (1.f + __expf(-v)));
}

// ---------------------------------------------------------------------------
// WMMA GEMM:  C[M,N] = A[M,K](f32) * W[N,K]^T(bf16)  (+bias[n]) (+add[m,n]) ; op
// op: 0=none 1=silu 2=softplus
// One wave computes one 16x16 tile with v_wmma_f32_16x16x32_bf16.
// Fragment layouts per CDNA5 ISA 7.12.2:
//   A (16x32 bf16): lanes 0-15 -> M=lane, K={0..7,16..23}; lanes 16-31 -> K+8
//   B mirrors A with N in place of M (weights stored N x K row-major)
//   C/D f32: elem r -> M = r + (lane<16?0:8), N = lane&15
// Alignment contract (guaranteed by callers): A rows 16B-aligned (lda%4==0,
// base 256B-aligned); W rows 16B-aligned whenever K%32==0 (fast path only).
// ---------------------------------------------------------------------------
__global__ void k_gemm(const float* __restrict__ A, int lda,
                       const bf16_t* __restrict__ W, int ldw,
                       float* __restrict__ C, int ldc,
                       const float* __restrict__ bias,
                       const float* __restrict__ addsrc, int ldadd,
                       int M, int N, int K, int op)
{
  const int lane = threadIdx.x;
  const int tn = (blockIdx.x * blockDim.y + threadIdx.y) * 16;
  const int tm = blockIdx.y * 16;
  if (tn >= N) return;                       // wave-uniform: EXEC stays all-1
  const int l15  = lane & 15;
  const int half = lane >> 4;
  const int koff = half * 8;
  int arow = tm + l15; if (arow > M - 1) arow = M - 1;   // clamp, store-guarded
  int brow = tn + l15; if (brow > N - 1) brow = N - 1;
  const float*  Ap = A + (size_t)arow * lda;
  const bf16_t* Bp = W + (size_t)brow * ldw;

  v8f acc = {0.f, 0.f, 0.f, 0.f, 0.f, 0.f, 0.f, 0.f};

  const int Kfull = K & ~31;
  int k0 = 0;
  // ---- fast path: whole 32-wide K slabs, unconditional 16B vector loads ----
  for (; k0 < Kfull; k0 += 32) {
    const float4 a0 = *(const float4*)(Ap + k0 + koff);
    const float4 a1 = *(const float4*)(Ap + k0 + koff + 4);
    const float4 a2 = *(const float4*)(Ap + k0 + koff + 16);
    const float4 a3 = *(const float4*)(Ap + k0 + koff + 20);
    const v8bf  b0 = *(const v8bf*)(Bp + k0 + koff);
    const v8bf  b1 = *(const v8bf*)(Bp + k0 + koff + 16);
    v16bf af, bfm;
    af[0] = (bf16_t)a0.x; af[1] = (bf16_t)a0.y; af[2]  = (bf16_t)a0.z; af[3]  = (bf16_t)a0.w;
    af[4] = (bf16_t)a1.x; af[5] = (bf16_t)a1.y; af[6]  = (bf16_t)a1.z; af[7]  = (bf16_t)a1.w;
    af[8] = (bf16_t)a2.x; af[9] = (bf16_t)a2.y; af[10] = (bf16_t)a2.z; af[11] = (bf16_t)a2.w;
    af[12] = (bf16_t)a3.x; af[13] = (bf16_t)a3.y; af[14] = (bf16_t)a3.z; af[15] = (bf16_t)a3.w;
#pragma unroll
    for (int j = 0; j < 8; ++j) { bfm[j] = b0[j]; bfm[j + 8] = b1[j]; }
    acc = __builtin_amdgcn_wmma_f32_16x16x32_bf16(false, af, false, bfm,
                                                  (short)0, acc, false, false);
    __builtin_prefetch(Ap + k0 + 32, 0, 1);   // global_prefetch_b8 (speculative)
    __builtin_prefetch(Bp + k0 + 32, 0, 1);
  }
  // ---- tail (K % 32 != 0, only dt_proj r in {4,8,16}): branchless selects ----
  if (k0 < K) {
    v16bf af, bfm;
#pragma unroll
    for (int j = 0; j < 8; ++j) {
      const int ka = k0 + koff + j;
      const int kb = ka + 16;
      const int kac = (ka < K) ? ka : (K - 1);
      const int kbc = (kb < K) ? kb : (K - 1);
      const float   av = Ap[kac], bv = Ap[kbc];
      const bf16_t  aw = Bp[kac], bw = Bp[kbc];
      af[j]      = (ka < K) ? (bf16_t)av : (bf16_t)0.f;
      af[j + 8]  = (kb < K) ? (bf16_t)bv : (bf16_t)0.f;
      bfm[j]     = (ka < K) ? aw : (bf16_t)0.f;
      bfm[j + 8] = (kb < K) ? bw : (bf16_t)0.f;
    }
    acc = __builtin_amdgcn_wmma_f32_16x16x32_bf16(false, af, false, bfm,
                                                  (short)0, acc, false, false);
  }

  // ---- epilogue ----
  const int n = tn + l15;
  float bv = 0.f;
  if (bias) bv = bias[(n < N) ? n : (N - 1)];
  const bool fullTile = (tm + 16 <= M) && (tn + 16 <= N);   // wave-uniform
  if (fullTile) {
#pragma unroll
    for (int r = 0; r < 8; ++r) {
      const int m = tm + r + half * 8;
      float v = acc[r] + bv;
      if (addsrc) v += addsrc[(size_t)m * ldadd + n];
      if (op == 1)      v = siluf(v);
      else if (op == 2) v = (v > 20.f) ? v : log1pf(__expf(v));
      C[(size_t)m * ldc + n] = v;
    }
  } else {
#pragma unroll
    for (int r = 0; r < 8; ++r) {
      const int m = tm + r + half * 8;
      if (m < M && n < N) {
        float v = acc[r] + bv;
        if (addsrc) v += addsrc[(size_t)m * ldadd + n];
        if (op == 1)      v = siluf(v);
        else if (op == 2) v = (v > 20.f) ? v : log1pf(__expf(v));
        C[(size_t)m * ldc + n] = v;
      }
    }
  }
}

// ---------------- weight preparation ----------------
__global__ void k_castw(const float* __restrict__ src, bf16_t* __restrict__ dst, int n) {
  int i = blockIdx.x * blockDim.x + threadIdx.x;
  if (i < n) dst[i] = (bf16_t)src[i];
}

// weight-norm: w[o][i] = g[o] * v[o][i] * rsqrt(sum_i v^2)   -> bf16 N x K
__global__ void k_wnorm(const float* __restrict__ v, const float* __restrict__ g,
                        bf16_t* __restrict__ dst, int I) {
  __shared__ float red[256];
  const int o = blockIdx.x, t = threadIdx.x;
  float s = 0.f;
  for (int i = t; i < I; i += blockDim.x) { float x = v[(size_t)o * I + i]; s += x * x; }
  red[t] = s; __syncthreads();
  for (int w = 128; w > 0; w >>= 1) { if (t < w) red[t] += red[t + w]; __syncthreads(); }
  const float scale = g[o] * rsqrtf(red[0]);
  for (int i = t; i < I; i += blockDim.x)
    dst[(size_t)o * I + i] = (bf16_t)(v[(size_t)o * I + i] * scale);
}

// ---------------- elementwise / norm / conv / scan ----------------
__global__ void k_rmsnorm(const float* __restrict__ x, const float* __restrict__ w,
                          float* __restrict__ out, int d) {
  __shared__ float red[256];
  const int row = blockIdx.x, t = threadIdx.x;
  const float* xr = x + (size_t)row * d;
  float s = 0.f;
  for (int c = t; c < d; c += blockDim.x) { float v = xr[c]; s += v * v; }
  red[t] = s; __syncthreads();
  for (int w2 = 128; w2 > 0; w2 >>= 1) { if (t < w2) red[t] += red[t + w2]; __syncthreads(); }
  const float scale = rsqrtf(red[0] / (float)d + 1e-5f);
  for (int c = t; c < d; c += blockDim.x) out[(size_t)row * d + c] = xr[c] * scale * w[c];
}

// causal depthwise conv (D_CONV=4) + bias + silu; input = xh half of xz
__global__ void k_dwconv_silu(const float* __restrict__ xz, int ldxz,
                              const float* __restrict__ cw, const float* __restrict__ cb,
                              float* __restrict__ out, int ed) {
  const int idx = blockIdx.x * blockDim.x + threadIdx.x;
  if (idx >= M_ * ed) return;
  const int e = idx % ed, m = idx / ed, l = m % L_;
  float acc = cb[e];
#pragma unroll
  for (int t = 0; t < 4; ++t) {
    const int ll = l - 3 + t;
    if (ll >= 0) acc += cw[e * 4 + t] * xz[(size_t)(m - 3 + t) * ldxz + e];
  }
  out[idx] = siluf(acc);
}

// selective scan: one thread per (b, e); 16 states in registers
__global__ void k_scan(const float* __restrict__ delta, const float* __restrict__ xh,
                       const float* __restrict__ dbc, int lddbc, int roff,
                       const float* __restrict__ A_log, const float* __restrict__ Dp,
                       float* __restrict__ y, int ed) {
  const int idx = blockIdx.x * blockDim.x + threadIdx.x;
  if (idx >= B_ * ed) return;
  const int e = idx % ed, b = idx / ed;
  float a[NSTATE], h[NSTATE];
#pragma unroll
  for (int n = 0; n < NSTATE; ++n) { a[n] = -__expf(A_log[e * NSTATE + n]); h[n] = 0.f; }
  const float dcoef = Dp[e];
  for (int t = 0; t < L_; ++t) {
    const int m  = b * L_ + t;
    const float dt = delta[(size_t)m * ed + e];
    const float xv = xh[(size_t)m * ed + e];
    const float dx = dt * xv;
    const float* bc = dbc + (size_t)m * lddbc + roff;
    float acc = 0.f;
#pragma unroll
    for (int n = 0; n < NSTATE; ++n) {
      h[n] = __expf(dt * a[n]) * h[n] + dx * bc[n];
      acc += h[n] * bc[16 + n];
    }
    y[(size_t)m * ed + e] = acc + dcoef * xv;
  }
}

__global__ void k_gate(float* __restrict__ y, const float* __restrict__ xz, int ldxz, int ed) {
  const int idx = blockIdx.x * blockDim.x + threadIdx.x;
  if (idx >= M_ * ed) return;
  const int e = idx % ed, m = idx / ed;
  y[idx] *= siluf(xz[(size_t)m * ldxz + ed + e]);
}

__global__ void k_add(float* __restrict__ a, const float* __restrict__ b, int n) {
  const int i = blockIdx.x * blockDim.x + threadIdx.x;
  if (i < n) a[i] += b[i];
}

__global__ void k_addrow(float* __restrict__ h, const float* __restrict__ de, int C) {
  const int i = blockIdx.x * blockDim.x + threadIdx.x;
  if (i >= M_ * C) return;
  const int c = i % C, m = i / C, b = m / L_;
  h[i] += de[b * C + c];
}

// (B,C,L) -> (B*L, C)
__global__ void k_ncl_to_lc(const float* __restrict__ src, float* __restrict__ dst, int C) {
  const int i = blockIdx.x * blockDim.x + threadIdx.x;
  if (i >= B_ * C * L_) return;
  const int l = i % L_, t = i / L_, ch = t % C, b = t / C;
  dst[((size_t)(b * L_ + l)) * C + ch] = src[i];
}

__global__ void k_outgate(const float* __restrict__ h, float* __restrict__ o) {
  const int i = blockIdx.x * blockDim.x + threadIdx.x;
  if (i >= M_ * RES_CH) return;
  const int c = i % RES_CH, m = i / RES_CH;
  const float a = tanhf(h[(size_t)m * 512 + c]);
  const float s = 1.f / (1.f + __expf(-h[(size_t)m * 512 + 256 + c]));
  o[i] = a * s;
}

__global__ void k_out_res(const float* __restrict__ xin, const float* __restrict__ res,
                          float* __restrict__ out) {
  const int i = blockIdx.x * blockDim.x + threadIdx.x;
  if (i >= B_ * IN_CH * L_) return;
  const int l = i % L_, t = i / L_, ch = t % IN_CH, b = t / IN_CH;
  out[i] = (xin[i] + res[((size_t)(b * L_ + l)) * IN_CH + ch]) * SQRT_HALF;
}

__global__ void k_out_skip(const float* __restrict__ skip, float* __restrict__ out) {
  const int i = blockIdx.x * blockDim.x + threadIdx.x;
  if (i >= B_ * RES_CH * L_) return;
  const int l = i % L_, t = i / L_, ch = t % RES_CH, b = t / RES_CH;
  out[i] = skip[((size_t)(b * L_ + l)) * RES_CH + ch];
}

// ---------------- host-side parameter views & context ----------------
struct ConvP  { const float *b, *g, *v; int O, I; };
struct MambaP { const float *A_log, *D, *conv_b, *conv_w, *dt_b, *dt_w,
                            *in_proj, *norm, *out_proj, *x_proj; int d, ed, r; };
struct UrmP   { ConvP dconv[2]; MambaP dm[2]; MambaP mid; ConvP uconv[2]; MambaP um[2]; };

struct Ctx {
  hipStream_t st;
  float *actA, *actB, *actC, *actD, *s0, *s1;
  float *x64, *cond128, *deb;
  float *xz, *xh, *dbc, *delta, *ybuf, *nrm;
  float *gateO, *res64, *skipO;
  bf16_t *wbuf;
};

static inline int ceil_div(int a, int b) { return (a + b - 1) / b; }

static void gemm(hipStream_t st, const float* A, int lda, const bf16_t* W, int ldw,
                 float* C, int ldc, const float* bias, const float* add, int ldadd,
                 int M, int N, int K, int op) {
  dim3 blk(32, 4, 1);
  dim3 grd(ceil_div(N, 64), ceil_div(M, 16), 1);
  k_gemm<<<grd, blk, 0, st>>>(A, lda, W, ldw, C, ldc, bias, add, ldadd, M, N, K, op);
}

static void castw(Ctx& c, const float* src, int n) {
  k_castw<<<ceil_div(n, 256), 256, 0, c.st>>>(src, c.wbuf, n);
}

static void conv1x1(Ctx& c, const float* x, float* out, const ConvP& p) {
  k_wnorm<<<p.O, 256, 0, c.st>>>(p.v, p.g, c.wbuf, p.I);
  gemm(c.st, x, p.I, c.wbuf, p.I, out, p.O, p.b, nullptr, 0, M_, p.O, p.I, 0);
}

static void mamba_layer(Ctx& c, float* x, const MambaP& p) {
  const int d = p.d, ed = p.ed, r = p.r;
  constexpr int LDDBC = 64;  // padded so dbc rows stay 16B-aligned for k_gemm
  k_rmsnorm<<<M_, 256, 0, c.st>>>(x, p.norm, c.nrm, d);
  castw(c, p.in_proj, 2 * ed * d);
  gemm(c.st, c.nrm, d, c.wbuf, d, c.xz, 2 * ed, nullptr, nullptr, 0, M_, 2 * ed, d, 0);
  { int tot = M_ * ed;
    k_dwconv_silu<<<ceil_div(tot, 256), 256, 0, c.st>>>(c.xz, 2 * ed, p.conv_w, p.conv_b, c.xh, ed); }
  castw(c, p.x_proj, (r + 2 * NSTATE) * ed);
  gemm(c.st, c.xh, ed, c.wbuf, ed, c.dbc, LDDBC, nullptr, nullptr, 0, M_, r + 32, ed, 0);
  castw(c, p.dt_w, ed * r);
  gemm(c.st, c.dbc, LDDBC, c.wbuf, r, c.delta, ed, p.dt_b, nullptr, 0, M_, ed, r, 2);
  { int tot = B_ * ed;
    k_scan<<<ceil_div(tot, 256), 256, 0, c.st>>>(c.delta, c.xh, c.dbc, LDDBC, r,
                                                 p.A_log, p.D, c.ybuf, ed); }
  { int tot = M_ * ed;
    k_gate<<<ceil_div(tot, 256), 256, 0, c.st>>>(c.ybuf, c.xz, 2 * ed, ed); }
  castw(c, p.out_proj, d * ed);
  gemm(c.st, c.ybuf, ed, c.wbuf, ed, x, d, nullptr, x, d, M_, d, ed, 0);  // +residual
}

static void run_urm(Ctx& c, float* x, float* t1, float* t2, const UrmP& u, int d) {
  hipMemcpyAsync(c.s0, x, (size_t)M_ * d * sizeof(float), hipMemcpyDeviceToDevice, c.st);
  conv1x1(c, x, t1, u.dconv[0]);  mamba_layer(c, t1, u.dm[0]);
  hipMemcpyAsync(c.s1, t1, (size_t)M_ * (d / 2) * sizeof(float), hipMemcpyDeviceToDevice, c.st);
  conv1x1(c, t1, t2, u.dconv[1]); mamba_layer(c, t2, u.dm[1]);
  mamba_layer(c, t2, u.mid);
  conv1x1(c, t2, t1, u.uconv[0]); mamba_layer(c, t1, u.um[0]);
  { int n = M_ * (d / 2); k_add<<<ceil_div(n, 256), 256, 0, c.st>>>(t1, c.s1, n); }
  conv1x1(c, t1, x, u.uconv[1]);  mamba_layer(c, x, u.um[1]);
  { int n = M_ * d;       k_add<<<ceil_div(n, 256), 256, 0, c.st>>>(x, c.s0, n); }
}

} // namespace

// ---------------------------------------------------------------------------
extern "C" void kernel_launch(void* const* d_in, const int* in_sizes, int n_in,
                              void* d_out, int out_size, void* d_ws, size_t ws_size,
                              hipStream_t stream) {
  (void)in_sizes; (void)n_in; (void)out_size; (void)ws_size;

  // -------- param tree walk (jax pytree: dict keys sorted alphabetically) ----
  int cur = 3;  // 0:x 1:cond 2:diffusion_step_embed, then params leaves
  auto F = [&](int i) { return (const float*)d_in[i]; };
  auto takeConv = [&](int O, int I) {
    ConvP q; q.b = F(cur++); q.g = F(cur++); q.v = F(cur++); q.O = O; q.I = I; return q; };
  auto takeMamba = [&](int d) {   // A_log,D,conv_b,conv_w,dt_proj_b,dt_proj_w,in_proj,norm,out_proj,x_proj
    MambaP q;
    q.A_log = F(cur++); q.D = F(cur++); q.conv_b = F(cur++); q.conv_w = F(cur++);
    q.dt_b = F(cur++); q.dt_w = F(cur++); q.in_proj = F(cur++); q.norm = F(cur++);
    q.out_proj = F(cur++); q.x_proj = F(cur++);
    q.d = d; q.ed = 2 * d; q.r = (d + 15) / 16; return q; };
  auto takeUrm = [&](int d) {     // keys: downs, mid, ups
    UrmP u;
    u.dconv[0] = takeConv(d / 2, d);     u.dm[0] = takeMamba(d / 2);
    u.dconv[1] = takeConv(d / 4, d / 2); u.dm[1] = takeMamba(d / 4);
    u.mid      = takeMamba(d / 4);
    u.uconv[0] = takeConv(d / 2, d / 4); u.um[0] = takeMamba(d / 2);
    u.uconv[1] = takeConv(d, d / 2);     u.um[1] = takeMamba(d);
    return u; };

  // top-level params, alphabetical:
  ConvP  cond_conv = takeConv(2 * RES_CH, 2 * IN_CH);          // (512,128)
  MambaP cond_ssm0 = takeMamba(512), cond_ssm1 = takeMamba(512);
  const float* diff_b = F(cur++);
  const float* diff_w = F(cur++);                              // (256,128)
  ConvP  input_proj = takeConv(256, IN_CH);
  MambaP input_ssm  = takeMamba(256);
  ConvP  mid_projection = takeConv(512, 256);
  ConvP  res_conv  = takeConv(IN_CH, RES_CH);
  ConvP  skip_conv = takeConv(RES_CH, RES_CH);
  UrmP   urm1 = takeUrm(256);
  UrmP   urm2 = takeUrm(512);

  // -------- workspace arena --------
  char* base = (char*)d_ws; size_t off = 0;
  auto takeF = [&](size_t n) { off = (off + 255) & ~size_t(255);
                               float* p = (float*)(base + off); off += n * 4; return p; };
  Ctx c; c.st = stream;
  c.actA = takeF((size_t)M_ * 512); c.actB = takeF((size_t)M_ * 512);
  c.actC = takeF((size_t)M_ * 512); c.actD = takeF((size_t)M_ * 512);
  c.s0 = takeF((size_t)M_ * 512);   c.s1 = takeF((size_t)M_ * 256);
  c.x64 = takeF((size_t)M_ * 64);   c.cond128 = takeF((size_t)M_ * 128);
  c.deb = takeF(2 * 256);
  c.xz = takeF((size_t)M_ * 2048);  c.xh = takeF((size_t)M_ * 1024);
  c.dbc = takeF((size_t)M_ * 64);   c.delta = takeF((size_t)M_ * 1024);
  c.ybuf = takeF((size_t)M_ * 1024); c.nrm = takeF((size_t)M_ * 512);
  c.gateO = takeF((size_t)M_ * 256); c.res64 = takeF((size_t)M_ * 64);
  c.skipO = takeF((size_t)M_ * 256);
  off = (off + 255) & ~size_t(255);
  c.wbuf = (bf16_t*)(base + off);   off += (size_t)2 * 1024 * 1024 * sizeof(bf16_t);

  const float* xin   = F(0);
  const float* cond  = F(1);
  const float* demb  = F(2);      // (2,128)
  float* dout = (float*)d_out;

  // -------- forward --------
  { int n = B_ * IN_CH * L_;      k_ncl_to_lc<<<ceil_div(n,256),256,0,stream>>>(xin,  c.x64,    IN_CH); }
  conv1x1(c, c.x64, c.actA, input_proj);                 // h : M x 256
  mamba_layer(c, c.actA, input_ssm);
  castw(c, diff_w, 256 * DIFF_DIM);
  gemm(stream, demb, DIFF_DIM, c.wbuf, DIFF_DIM, c.deb, 256, diff_b, nullptr, 0,
       B_, 256, DIFF_DIM, 0);                            // de : 2 x 256
  { int n = M_ * 256;             k_addrow<<<ceil_div(n,256),256,0,stream>>>(c.actA, c.deb, 256); }
  run_urm(c, c.actA, c.actB, c.actC, urm1, 256);
  conv1x1(c, c.actA, c.actB, mid_projection);            // h : M x 512 in actB
  { int n = B_ * 2 * IN_CH * L_;  k_ncl_to_lc<<<ceil_div(n,256),256,0,stream>>>(cond, c.cond128, 2*IN_CH); }
  conv1x1(c, c.cond128, c.actC, cond_conv);              // c : M x 512
  mamba_layer(c, c.actC, cond_ssm0);
  mamba_layer(c, c.actC, cond_ssm1);
  { int n = M_ * 512;             k_add<<<ceil_div(n,256),256,0,stream>>>(c.actB, c.actC, n); }
  run_urm(c, c.actB, c.actC, c.actD, urm2, 512);
  { int n = M_ * RES_CH;          k_outgate<<<ceil_div(n,256),256,0,stream>>>(c.actB, c.gateO); }
  conv1x1(c, c.gateO, c.res64, res_conv);
  conv1x1(c, c.gateO, c.skipO, skip_conv);
  { int n = B_ * IN_CH * L_;      k_out_res<<<ceil_div(n,256),256,0,stream>>>(xin, c.res64, dout); }
  { int n = B_ * RES_CH * L_;     k_out_skip<<<ceil_div(n,256),256,0,stream>>>(c.skipO, dout + (size_t)B_*IN_CH*L_); }
}
